// MambaBlock_82523501625677
// MI455X (gfx1250) — matrix operand order repaired and checked by reading, not compiled, not run on hardware
//
#include <hip/hip_runtime.h>
#include <cstdint>
#include <cstddef>

// ---------------------------------------------------------------------------
// Mamba block forward for MI455X (gfx1250, wave32, WMMA).
// GEMMs: v_wmma_f32_16x16x32_bf16, branch-free double-buffered inner loop,
// wave tile 32x64 (8 WMMA per 12 b128 loads). Norm/conv/softplus/scan in fp32.
// ---------------------------------------------------------------------------

typedef __attribute__((ext_vector_type(16))) __bf16 v16bf;
typedef __attribute__((ext_vector_type(8)))  __bf16 v8bf;
typedef __attribute__((ext_vector_type(8)))  float  v8f;

#define LSEQ    2048
#define DMODEL  1024
#define DINNER  2048
#define DSTATE  16
#define DTRANK  64

#define WMMA_BF16(a, b, c) \
  __builtin_amdgcn_wmma_f32_16x16x32_bf16(false, (a), false, (b), (short)0, (c), false, false)

static __device__ __forceinline__ __bf16 f2bf(float f) {
  union { float f; uint32_t u; } v; v.f = f;
  uint32_t r = v.u + 0x7fffu + ((v.u >> 16) & 1u);   // round-to-nearest-even
  union { uint16_t s; __bf16 b; } o; o.s = (uint16_t)(r >> 16);
  return o.b;
}

static __device__ __forceinline__ float siluf(float x) {
  return x / (1.f + __expf(-x));
}

static __device__ __forceinline__ int imin(int a, int b) { return a < b ? a : b; }

// A fragment (16x32 bf16): lane hl=lane>>4 holds K = {hl*16..hl*16+15}
// p already includes row*K + hl*16.
static __device__ __forceinline__ v16bf load_a(const __bf16* __restrict__ p, int k0) {
  const v8bf lo = *(const v8bf*)(p + k0);
  const v8bf hi = *(const v8bf*)(p + k0 + 8);
  return __builtin_shufflevector(lo, hi, 0,1,2,3,4,5,6,7,8,9,10,11,12,13,14,15);
}

// ------------------------------- GEMM kernel -------------------------------
// C[M,N] = A[M,K] @ Bt[N,K]^T (+bias[N]) (mode1: softplus) (+residual[M,N])
// Block: 256 threads = 8 waves (4 in M x 2 in N); wave tile 32x64;
// block tile 128x128. Requires K % 32 == 0 (true for 1024/2048/64).
// Out-of-range rows/cols are CLAMPED for loads (garbage never stored), so the
// inner loop is branch-free and EXEC stays all-ones for WMMA.
__global__ __launch_bounds__(256) void gemm_bf16_wmma(
    const __bf16* __restrict__ A, const __bf16* __restrict__ Bt,
    float* __restrict__ C, const float* __restrict__ bias,
    const float* __restrict__ residual, int M, int N, int K, int mode) {
  const int lane = threadIdx.x & 31;
  const int wave = threadIdx.x >> 5;
  const int hl   = lane >> 4;
  const int r    = lane & 15;
  const int wm   = wave & 3;      // 0..3  (M direction)
  const int wn   = wave >> 2;     // 0..1  (N direction)
  const int mbase = blockIdx.y * 128 + wm * 32;
  const int nbase = blockIdx.x * 128 + wn * 64;

  // Clamped base pointers (lane K-offset folded in).
  const __bf16* pA0 = A + (size_t)imin(mbase + r,      M - 1) * K + hl * 16;
  const __bf16* pA1 = A + (size_t)imin(mbase + 16 + r, M - 1) * K + hl * 16;
  const __bf16* pB0 = Bt + (size_t)imin(nbase + r,      N - 1) * K + hl * 16;
  const __bf16* pB1 = Bt + (size_t)imin(nbase + 16 + r, N - 1) * K + hl * 16;
  const __bf16* pB2 = Bt + (size_t)imin(nbase + 32 + r, N - 1) * K + hl * 16;
  const __bf16* pB3 = Bt + (size_t)imin(nbase + 48 + r, N - 1) * K + hl * 16;

  v8f c00 = {}, c01 = {}, c02 = {}, c03 = {};
  v8f c10 = {}, c11 = {}, c12 = {}, c13 = {};

  // Prologue loads.
  v16bf a0 = load_a(pA0, 0);
  v16bf a1 = load_a(pA1, 0);
  v16bf b0 = *(const v16bf*)pB0;
  v16bf b1 = *(const v16bf*)pB1;
  v16bf b2 = *(const v16bf*)pB2;
  v16bf b3 = *(const v16bf*)pB3;

  for (int k0 = 32; k0 < K; k0 += 32) {
    // Prefetch next k-step while current WMMAs execute.
    v16bf na0 = load_a(pA0, k0);
    v16bf na1 = load_a(pA1, k0);
    v16bf nb0 = *(const v16bf*)(pB0 + k0);
    v16bf nb1 = *(const v16bf*)(pB1 + k0);
    v16bf nb2 = *(const v16bf*)(pB2 + k0);
    v16bf nb3 = *(const v16bf*)(pB3 + k0);

    c00 = WMMA_BF16(a0, b0, c00);
    c01 = WMMA_BF16(a0, b1, c01);
    c02 = WMMA_BF16(a0, b2, c02);
    c03 = WMMA_BF16(a0, b3, c03);
    c10 = WMMA_BF16(a1, b0, c10);
    c11 = WMMA_BF16(a1, b1, c11);
    c12 = WMMA_BF16(a1, b2, c12);
    c13 = WMMA_BF16(a1, b3, c13);

    a0 = na0; a1 = na1;
    b0 = nb0; b1 = nb1; b2 = nb2; b3 = nb3;
  }
  // Tail k-step.
  c00 = WMMA_BF16(a0, b0, c00);
  c01 = WMMA_BF16(a0, b1, c01);
  c02 = WMMA_BF16(a0, b2, c02);
  c03 = WMMA_BF16(a0, b3, c03);
  c10 = WMMA_BF16(a1, b0, c10);
  c11 = WMMA_BF16(a1, b1, c11);
  c12 = WMMA_BF16(a1, b2, c12);
  c13 = WMMA_BF16(a1, b3, c13);

  // C/D layout: v8f elem i -> row = tileM + i + hl*8, col = tileN + r.
  v8f accs[2][4] = {{c00, c01, c02, c03}, {c10, c11, c12, c13}};
#pragma unroll
  for (int jn = 0; jn < 4; ++jn) {
    const int col = nbase + jn * 16 + r;
    if (col >= N) continue;
    const float bval = bias ? bias[col] : 0.f;
#pragma unroll
    for (int im = 0; im < 2; ++im) {
      const int rowb = mbase + im * 16 + hl * 8;
      if (rowb >= M) continue;
#pragma unroll
      for (int i = 0; i < 8; ++i) {
        const int row = rowb + i;
        if (row >= M) continue;
        float v = accs[im][jn][i] + bval;
        if (mode == 1) v = (v > 20.f) ? v : log1pf(__expf(v));   // softplus
        if (residual)  v += residual[(size_t)row * N + col];
        C[(size_t)row * N + col] = v;
      }
    }
  }
}

// ------------------------------ RMSNorm -> bf16 ----------------------------
__global__ __launch_bounds__(256) void rmsnorm_to_bf16(
    const float* __restrict__ x, const float* __restrict__ scale,
    __bf16* __restrict__ u) {
  __shared__ float red[256];
  const int row = blockIdx.x;
  const float* xr = x + (size_t)row * DMODEL;
  float s = 0.f;
  for (int c = threadIdx.x; c < DMODEL; c += 256) { float v = xr[c]; s += v * v; }
  red[threadIdx.x] = s;
  __syncthreads();
  for (int off = 128; off > 0; off >>= 1) {
    if ((int)threadIdx.x < off) red[threadIdx.x] += red[threadIdx.x + off];
    __syncthreads();
  }
  const float inv = rsqrtf(red[0] / (float)DMODEL + 1e-6f);
  for (int c = threadIdx.x; c < DMODEL; c += 256)
    u[(size_t)row * DMODEL + c] = f2bf(xr[c] * inv * scale[c]);
}

// --------------------- weight transpose (K,N)f32 -> (N,K)bf16 --------------
__global__ __launch_bounds__(256) void transpose_to_bf16(
    const float* __restrict__ in, __bf16* __restrict__ out, int K, int N) {
  int id = blockIdx.x * 256 + threadIdx.x;
  if (id >= K * N) return;
  int n = id / K, k = id - n * K;
  out[id] = f2bf(in[(size_t)k * N + n]);
}

// --------- depthwise conv along d (per the reference's transpose) + SiLU ---
// x_new[l,d] = silu( conv_b[l] + sum_k x[l, d-1+k] * conv_w[k,0,l] )
__global__ __launch_bounds__(256) void dconv_silu(
    const float* __restrict__ xz, const float* __restrict__ conv_w,
    const float* __restrict__ conv_b, float* __restrict__ xc) {
  int id = blockIdx.x * 256 + threadIdx.x;
  if (id >= LSEQ * DINNER) return;
  int l = id / DINNER, d = id - l * DINNER;
  const float* row = xz + (size_t)l * (2 * DINNER);   // x = first DINNER cols
  float s = conv_b[l];
#pragma unroll
  for (int k = 0; k < 4; ++k) {
    int dd = d - 1 + k;                                // SAME pad: lo=1, hi=2
    if (dd >= 0 && dd < DINNER) s += row[dd] * conv_w[k * LSEQ + l];
  }
  xc[id] = siluf(s);
}

// -------- slice x_dbl[:, 0:64] -> bf16 for the dt GEMM ---------------------
__global__ __launch_bounds__(256) void slice_delta_bf16(
    const float* __restrict__ xdbl, __bf16* __restrict__ dlt) {
  int id = blockIdx.x * 256 + threadIdx.x;
  if (id >= LSEQ * DTRANK) return;
  int m = id / DTRANK, c = id - m * DTRANK;
  dlt[id] = f2bf(xdbl[(size_t)m * 96 + c]);
}

// ------------------------------ selective scan -----------------------------
// One channel per lane. h[16] in registers; B/C (32 floats per t) staged in
// LDS in 128-step chunks. Fuses +x*D, *silu(res), bf16 convert.
#define SCAN_T     64
#define SCAN_CHUNK 128
__global__ __launch_bounds__(SCAN_T) void selective_scan(
    const float* __restrict__ delta, const float* __restrict__ xc,
    const float* __restrict__ xz,    const float* __restrict__ xdbl,
    const float* __restrict__ A_log, const float* __restrict__ Dvec,
    __bf16* __restrict__ ybf) {
  __shared__ float sBC[SCAN_CHUNK * 32];
  const int d = blockIdx.x * SCAN_T + threadIdx.x;

  float Arow[DSTATE];
#pragma unroll
  for (int n = 0; n < DSTATE; ++n) Arow[n] = -__expf(A_log[(size_t)d * DSTATE + n]);
  const float Dd = Dvec[d];
  float h[DSTATE];
#pragma unroll
  for (int n = 0; n < DSTATE; ++n) h[n] = 0.f;

  for (int t0 = 0; t0 < LSEQ; t0 += SCAN_CHUNK) {
    __syncthreads();
    for (int idx = threadIdx.x; idx < SCAN_CHUNK * 32; idx += SCAN_T) {
      int tt = idx >> 5, c = idx & 31;
      sBC[idx] = xdbl[(size_t)(t0 + tt) * 96 + 64 + c];   // cols 64..95 = B,C
    }
    __syncthreads();
    for (int tt = 0; tt < SCAN_CHUNK; ++tt) {
      const int t = t0 + tt;
      const float dv = delta[(size_t)t * DINNER + d];
      const float xv = xc[(size_t)t * DINNER + d];
      const float rv = xz[(size_t)t * (2 * DINNER) + DINNER + d];
      const float* bc = &sBC[tt * 32];
      float acc = 0.f;
#pragma unroll
      for (int n = 0; n < DSTATE; ++n) {
        const float dA = __expf(dv * Arow[n]);
        h[n] = dA * h[n] + (dv * bc[n]) * xv;
        acc += h[n] * bc[16 + n];
      }
      float y = acc + xv * Dd;
      y *= siluf(rv);
      ybf[(size_t)t * DINNER + d] = f2bf(y);
    }
  }
}

// ------------------------------- launcher ----------------------------------
extern "C" void kernel_launch(void* const* d_in, const int* in_sizes, int n_in,
                              void* d_out, int out_size, void* d_ws, size_t ws_size,
                              hipStream_t stream) {
  (void)in_sizes; (void)n_in; (void)out_size; (void)ws_size;
  const float* inputs = (const float*)d_in[0];
  const float* nscale = (const float*)d_in[1];
  const float* w_in   = (const float*)d_in[2];
  const float* conv_w = (const float*)d_in[3];
  const float* conv_b = (const float*)d_in[4];
  const float* w_xp   = (const float*)d_in[5];
  const float* w_dt   = (const float*)d_in[6];
  const float* b_dt   = (const float*)d_in[7];
  const float* w_out  = (const float*)d_in[8];
  const float* A_log  = (const float*)d_in[9];
  const float* Dvec   = (const float*)d_in[10];
  float* out = (float*)d_out;

  uint8_t* ws = (uint8_t*)d_ws;
  size_t off = 0;
  auto alloc = [&](size_t bytes) -> void* {
    void* p = ws + off;
    off = (off + bytes + 255) & ~(size_t)255;
    return p;
  };
  __bf16* u_bf   = (__bf16*)alloc((size_t)LSEQ * DMODEL * 2);        // u, bf16
  __bf16* w1T    = (__bf16*)alloc((size_t)(2*DINNER) * DMODEL * 2);  // in_proj^T
  __bf16* wxT    = (__bf16*)alloc((size_t)96 * DMODEL * 2);          // x_proj^T
  __bf16* wdtT   = (__bf16*)alloc((size_t)DINNER * DTRANK * 2);      // dt_proj^T
  __bf16* woT    = (__bf16*)alloc((size_t)DMODEL * DINNER * 2);      // out_proj^T
  float*  xz     = (float*)alloc((size_t)LSEQ * 2*DINNER * 4);       // [x | res]
  float*  xc     = (float*)alloc((size_t)LSEQ * DINNER * 4);         // conv+silu
  float*  xdbl   = (float*)alloc((size_t)LSEQ * 96 * 4);             // delta|B|C
  __bf16* dlt_bf = (__bf16*)alloc((size_t)LSEQ * DTRANK * 2);
  float*  dfull  = (float*)alloc((size_t)LSEQ * DINNER * 4);         // softplus
  __bf16* y_bf   = (__bf16*)alloc((size_t)LSEQ * DINNER * 2);

  // 1) RMSNorm -> bf16 activations
  rmsnorm_to_bf16<<<LSEQ, 256, 0, stream>>>(inputs, nscale, u_bf);

  // 2) weights: f32 (K,N) -> bf16 (N,K) for contiguous-K WMMA B fragments
  transpose_to_bf16<<<(2*DINNER*DMODEL)/256, 256, 0, stream>>>(w_in, w1T, DMODEL, 2*DINNER);
  transpose_to_bf16<<<(96*DMODEL)/256,       256, 0, stream>>>(w_xp, wxT, DMODEL, 96);
  transpose_to_bf16<<<(DINNER*DTRANK)/256,   256, 0, stream>>>(w_dt, wdtT, DTRANK, DINNER);
  transpose_to_bf16<<<(DMODEL*DINNER)/256,   256, 0, stream>>>(w_out, woT, DINNER, DMODEL);

  // 3) xz = u @ in_proj  (2048x1024 @ 1024x4096)
  gemm_bf16_wmma<<<dim3((2*DINNER)/128, LSEQ/128), 256, 0, stream>>>(
      u_bf, w1T, xz, nullptr, nullptr, LSEQ, 2*DINNER, DMODEL, 0);

  // 4) depthwise conv + SiLU
  dconv_silu<<<(LSEQ*DINNER)/256, 256, 0, stream>>>(xz, conv_w, conv_b, xc);

  // 5) x_dbl = u @ x_proj  (N=96 -> one block column, store-guarded)
  gemm_bf16_wmma<<<dim3(1, LSEQ/128), 256, 0, stream>>>(
      u_bf, wxT, xdbl, nullptr, nullptr, LSEQ, 96, DMODEL, 0);

  // 6) delta = softplus(x_dbl[:, :64] @ dt_proj + b)  (K=64, fused epilogue)
  slice_delta_bf16<<<(LSEQ*DTRANK)/256, 256, 0, stream>>>(xdbl, dlt_bf);
  gemm_bf16_wmma<<<dim3(DINNER/128, LSEQ/128), 256, 0, stream>>>(
      dlt_bf, wdtT, dfull, b_dt, nullptr, LSEQ, DINNER, DTRANK, 1);

  // 7) selective scan (serial in L, 2048 channels across 64 waves)
  selective_scan<<<DINNER/SCAN_T, SCAN_T, 0, stream>>>(
      dfull, xc, xz, xdbl, A_log, Dvec, y_bf);

  // 8) out = y @ out_proj + inputs (residual fused into epilogue)
  gemm_bf16_wmma<<<dim3(DMODEL/128, LSEQ/128), 256, 0, stream>>>(
      y_bf, woT, out, nullptr, inputs, LSEQ, DMODEL, DINNER, 0);
}
